// MatmulLayer_17008070492685
// MI455X (gfx1250) — matrix operand order repaired and checked by reading, not compiled, not run on hardware
//
#include <hip/hip_runtime.h>
#include <stdint.h>

// W4A16 grouped-dequant GEMM for gfx1250 (MI455X).
// C[M,N](f32) = A[M,K](f16) x dequant(Q[K,N] int32, scales/zeros f16, group=128 in K)
//
// MI455X reasoning: Q is 64MB (int32 per 4-bit value) and dominates the ~160MB
// HBM traffic (~7us at 23.3 TB/s); FLOPs are 137G at the f16 WMMA ceiling.
// Strategy: stream Q once -> f16 dequant in registers -> v_wmma_f32_16x16x32_f16
// with f32 accum. A-tile is a pure copy, so move it with the Tensor Data Mover
// (tensor_load_to_lds, TENSORcnt) while VALU/VMEM handle the Q dequant stream.
// LDS is ping-pong double buffered: one s_barrier per K-step.

typedef __attribute__((ext_vector_type(16))) _Float16 v16h;
typedef __attribute__((ext_vector_type(8)))  _Float16 v8h;
typedef __attribute__((ext_vector_type(8)))  float    v8f;
typedef __attribute__((ext_vector_type(4)))  int      v4i;
typedef __attribute__((ext_vector_type(4)))  unsigned u32x4;
typedef __attribute__((ext_vector_type(8)))  int      i32x8;
typedef __attribute__((ext_vector_type(4)))  int      i32x4;

#define MDIM 4096
#define NDIM 4096
#define KDIM 4096

#define BM 128
#define BN 128
#define BK 32
#define LDSW (BK + 8)          // row stride in halves: 80 bytes (16B aligned)
#define NT (KDIM / BK)

#if defined(__has_builtin)
#  if __has_builtin(__builtin_amdgcn_tensor_load_to_lds) && \
      __has_builtin(__builtin_amdgcn_s_wait_tensorcnt)
#    define USE_TDM 1
#  endif
#endif
#ifndef USE_TDM
#  define USE_TDM 0
#endif

#if USE_TDM
__device__ __forceinline__ unsigned lds_byte_addr(const void* p) {
    // generic -> LDS(AS3) addrspacecast, then take the 32-bit LDS byte offset
    return (unsigned)(uintptr_t)(__attribute__((address_space(3))) const void*)p;
}

// Issue one TDM descriptor: copy a BM x BK tile of f16 A (row stride KDIM halves)
// into LDS with 64B data + 16B pad per row (matches LDSW = BK+8 halves).
__device__ __forceinline__ void tdm_issue_a(const _Float16* gsrc, unsigned ldsb) {
    unsigned long long ga = (unsigned long long)(uintptr_t)gsrc;
    u32x4 g0;
    g0[0] = 1u;                                            // count=1, user mode
    g0[1] = ldsb;                                          // lds_addr (bytes)
    g0[2] = (unsigned)ga;                                  // global_addr[31:0]
    g0[3] = (unsigned)((ga >> 32) & 0x01FFFFFFu) | (2u << 30); // addr[56:32], type=2
    i32x8 g1;
    g1[0] = (1 << 16)      // data_size = 1 -> 2 bytes
          | (1 << 20)      // pad_enable
          | (3 << 22)      // pad_interval: 16 DWORDs (64B) between pads
          | (3 << 25);     // pad_amount: 4 DWORDs (16B)
    g1[1] = (int)(((unsigned)KDIM & 0xFFFFu) << 16);       // tensor_dim0[15:0]
    g1[2] = (int)((((unsigned)KDIM >> 16) & 0xFFFFu) |
                  (((unsigned)MDIM & 0xFFFFu) << 16));     // dim0[31:16] | dim1[15:0]
    g1[3] = (int)((((unsigned)MDIM >> 16) & 0xFFFFu) |
                  ((unsigned)BK << 16));                   // dim1[31:16] | tile_dim0
    g1[4] = BM;                                            // tile_dim1 | tile_dim2=0
    g1[5] = KDIM;                                          // tensor_dim0_stride[31:0]
    g1[6] = 0;                                             // stride[47:32], dim1_stride lo
    g1[7] = 0;
    i32x4 z4 = (i32x4)0;
#if __clang_major__ >= 23
    i32x8 z8 = (i32x8)0;
    __builtin_amdgcn_tensor_load_to_lds(g0, g1, z4, z4, z8, 0);
#else
    __builtin_amdgcn_tensor_load_to_lds(g0, g1, z4, z4, 0);
#endif
}
#endif // USE_TDM

__device__ __forceinline__ void fetch_q(const int* __restrict__ Q,
                                        const _Float16* __restrict__ S,
                                        const _Float16* __restrict__ Z,
                                        int k0, int bn0, int w_n, int w_kc,
                                        int (&qv)[16], _Float16& sv, _Float16& zv) {
    const int g = k0 >> 7;                                 // GROUP=128, BK | GROUP
    sv = S[(size_t)g * NDIM + bn0 + w_n];
    zv = Z[(size_t)g * NDIM + bn0 + w_n];
    const int* qp = Q + (size_t)(k0 + w_kc) * NDIM + bn0 + w_n;
#pragma unroll
    for (int j = 0; j < 16; ++j) qv[j] = qp[(size_t)j * NDIM];
}

__device__ __forceinline__ void store_w(_Float16* ws, int w_n, int w_kc,
                                        const int (&qv)[16], _Float16 sv, _Float16 zv) {
    const _Float16 nzs = (_Float16)(-zv * sv);             // w = q*s + (-z*s)
    v8h w0, w1;
#pragma unroll
    for (int j = 0; j < 8; ++j) {
        w0[j] = (_Float16)(float)qv[j]     * sv + nzs;
        w1[j] = (_Float16)(float)qv[j + 8] * sv + nzs;
    }
    *(v8h*)(&ws[w_n * LDSW + w_kc])     = w0;              // k = w_kc .. +7
    *(v8h*)(&ws[w_n * LDSW + w_kc + 8]) = w1;              // k = w_kc+8 .. +15
}

__device__ __forceinline__ void mma_step(const _Float16* as, const _Float16* ws,
                                         int wm, int wn, int lane, v8f (&acc)[2][4]) {
    const int lrow = lane & 15;
    const int ah = (lane >> 4) << 3;   // A frag: split 8+8 K-runs (0 or 8)
    const int bh = (lane >> 4) << 3;   // B frag: split 8+8 K-runs (0 or 8)
    v16h afrag[2], bfrag[4];
#pragma unroll
    for (int mi = 0; mi < 2; ++mi) {
        const int r = wm * 32 + mi * 16 + lrow;
        union { v16h v; v8h h[2]; } u;
        u.h[0] = *(const v8h*)(&as[r * LDSW + ah]);        // K ah..+7
        u.h[1] = *(const v8h*)(&as[r * LDSW + ah + 16]);   // K ah+16..+23
        afrag[mi] = u.v;
    }
#pragma unroll
    for (int ni = 0; ni < 4; ++ni) {
        const int c = wn * 64 + ni * 16 + lrow;
        union { v16h v; v8h h[2]; } u;
        u.h[0] = *(const v8h*)(&ws[c * LDSW + bh]);        // K bh..+7
        u.h[1] = *(const v8h*)(&ws[c * LDSW + bh + 16]);   // K bh+16..+23
        bfrag[ni] = u.v;
    }
#pragma unroll
    for (int mi = 0; mi < 2; ++mi)
#pragma unroll
        for (int ni = 0; ni < 4; ++ni)
            acc[mi][ni] = __builtin_amdgcn_wmma_f32_16x16x32_f16(
                false, afrag[mi], false, bfrag[ni],
                (short)0, acc[mi][ni], false, false);
}

__global__ __launch_bounds__(256)
void w4a16_gemm_kernel(const _Float16* __restrict__ A,
                       const int*      __restrict__ Q,
                       const _Float16* __restrict__ S,
                       const _Float16* __restrict__ Z,
                       float*          __restrict__ C)
{
    __shared__ alignas(16) _Float16 As[2][BM * LDSW];
    __shared__ alignas(16) _Float16 Ws[2][BN * LDSW];      // transposed: Ws[n][k]

    const int tid  = threadIdx.x;
    const int bm0  = blockIdx.y * BM;
    const int bn0  = blockIdx.x * BN;

    const int wave = tid >> 5;          // 0..7
    const int lane = tid & 31;
    const int wm   = wave & 3;          // 4 waves down M: 32-row strips
    const int wn   = wave >> 2;         // 2 waves across N: 64-col strips

    // W loader: each thread owns column n = tid&127 and 16 k's (tid>>7 picks half)
    const int w_n  = tid & (BN - 1);
    const int w_kc = (tid >> 7) * 16;

#if !USE_TDM
    // Fallback A loader: row = tid/2, 16-half chunk = (tid&1)*16
    const int a_row = tid >> 1;
    const int a_c   = (tid & 1) * 16;
    v4i av0, av1;
#endif

    v8f acc[2][4];
#pragma unroll
    for (int i = 0; i < 2; ++i)
#pragma unroll
        for (int j = 0; j < 4; ++j)
            acc[i][j] = (v8f){0.f, 0.f, 0.f, 0.f, 0.f, 0.f, 0.f, 0.f};

    int qv[16];
    _Float16 sv, zv;

    // ---- prologue: stage tile 0 into buffer 0 ----
#if USE_TDM
    if (wave == 0)
        tdm_issue_a(A + (size_t)bm0 * KDIM, lds_byte_addr(&As[0][0]));
#else
    {
        const _Float16* srcA = A + (size_t)(bm0 + a_row) * KDIM + a_c;
        av0 = *(const v4i*)(srcA);
        av1 = *(const v4i*)(srcA + 8);
        *(v4i*)(&As[0][a_row * LDSW + a_c])     = av0;
        *(v4i*)(&As[0][a_row * LDSW + a_c + 8]) = av1;
    }
#endif
    fetch_q(Q, S, Z, /*k0=*/0, bn0, w_n, w_kc, qv, sv, zv);
    store_w(&Ws[0][0], w_n, w_kc, qv, sv, zv);
#if USE_TDM
    __builtin_amdgcn_s_wait_tensorcnt(0);
#endif
    __syncthreads();

    // ---- main pipeline: one barrier per K-step, ping-pong LDS ----
    int buf = 0;
    for (int kt = 0; kt < NT; ++kt) {
        const bool has_next = (kt + 1 < NT);
        if (has_next) {
            const int kn = (kt + 1) * BK;
#if USE_TDM
            if (wave == 0)
                tdm_issue_a(A + (size_t)bm0 * KDIM + kn, lds_byte_addr(&As[buf ^ 1][0]));
#else
            const _Float16* srcA = A + (size_t)(bm0 + a_row) * KDIM + kn + a_c;
            av0 = *(const v4i*)(srcA);
            av1 = *(const v4i*)(srcA + 8);
#endif
            fetch_q(Q, S, Z, kn, bn0, w_n, w_kc, qv, sv, zv);   // hide behind WMMA
        }

        mma_step(&As[buf][0], &Ws[buf][0], wm, wn, lane, acc);

        if (has_next) {
            store_w(&Ws[buf ^ 1][0], w_n, w_kc, qv, sv, zv);
#if !USE_TDM
            *(v4i*)(&As[buf ^ 1][a_row * LDSW + a_c])     = av0;
            *(v4i*)(&As[buf ^ 1][a_row * LDSW + a_c + 8]) = av1;
#endif
#if USE_TDM
            __builtin_amdgcn_s_wait_tensorcnt(0);
#endif
        }
        __syncthreads();
        buf ^= 1;
    }

    // ---- epilogue: C/D layout -> lanes 0-15 hold M=r, lanes 16-31 hold M=8+r ----
    const int lrow = lane & 15;
    const int rshift = (lane >> 4) << 3;
#pragma unroll
    for (int mi = 0; mi < 2; ++mi) {
        const int rbase = bm0 + wm * 32 + mi * 16 + rshift;
#pragma unroll
        for (int ni = 0; ni < 4; ++ni) {
            const int col = bn0 + wn * 64 + ni * 16 + lrow;
#pragma unroll
            for (int r = 0; r < 8; ++r) {
                C[(size_t)(rbase + r) * NDIM + col] = acc[mi][ni][r];
            }
        }
    }
}

extern "C" void kernel_launch(void* const* d_in, const int* in_sizes, int n_in,
                              void* d_out, int out_size, void* d_ws, size_t ws_size,
                              hipStream_t stream) {
    (void)in_sizes; (void)n_in; (void)out_size; (void)d_ws; (void)ws_size;
    const _Float16* A = (const _Float16*)d_in[0];
    const int*      Q = (const int*)d_in[1];
    const _Float16* S = (const _Float16*)d_in[2];
    const _Float16* Z = (const _Float16*)d_in[3];
    float*          C = (float*)d_out;

    dim3 grid(NDIM / BN, MDIM / BM);  // 32 x 32 blocks
    w4a16_gemm_kernel<<<grid, dim3(256), 0, stream>>>(A, Q, S, Z, C);
}
